// Enet_14766097563759
// MI455X (gfx1250) — matrix operand order repaired and checked by reading, not compiled, not run on hardware
//
#include <hip/hip_runtime.h>
#include <hip/hip_bf16.h>
#include <math.h>

// ---------------- problem constants (match reference) ----------------
#define NNODES   65536      // B * NPG
#define NEDGES   524288     // NNODES * 8
#define BGRAPH   64
#define NPGRAPH  1024
#define FNODE    128
#define FEDGE    6
#define DTOT     97         // 32+32+32+1
#define KTOP     30
#define C1N      16
#define C2N      32
#define KW2C     5
#define T2LEN    11         // K/2 - KW2 + 1
#define DENSEIN  352        // T2LEN * C2N
#define OUTDIM   128
#define HSTRIDE  160        // padded feature stride (multiple of 32)

typedef _Float16 v16h __attribute__((ext_vector_type(16)));
typedef _Float16 v8h  __attribute__((ext_vector_type(8)));
typedef float    v8f  __attribute__((ext_vector_type(8)));
typedef int      vint4 __attribute__((vector_size(16)));
typedef __attribute__((address_space(1))) vint4* gv4i_ptr;
typedef __attribute__((address_space(3))) vint4* lv4i_ptr;

// Opt-in async global->LDS staging (CDNA5); guarded so absence still compiles.
#ifndef USE_ASYNC_LDS
#define USE_ASYNC_LDS 1
#endif
#if USE_ASYNC_LDS && __has_builtin(__builtin_amdgcn_global_load_async_to_lds_b128) && \
    __has_builtin(__builtin_amdgcn_s_wait_asynccnt)
#define HAVE_ASYNC_LDS 1
#else
#define HAVE_ASYNC_LDS 0
#endif

// ---------------- workspace layout (bytes) ----------------
static const size_t OFF_PPR  = 0;
static const size_t OFF_E2N  = OFF_PPR  + (size_t)NEDGES * 4;                 // E f32
static const size_t OFF_H    = OFF_E2N  + (size_t)NNODES * FEDGE * 4;         // N*6 f32
static const size_t OFF_POOL = OFF_H    + (size_t)NNODES * HSTRIDE * 4;       // N*160 f32
static const size_t OFF_MSG  = OFF_POOL + (size_t)NNODES * HSTRIDE * 4;       // N*160 f32
static const size_t OFF_P16  = OFF_MSG  + (size_t)NNODES * DTOT * 4;          // N*97 f32
static const size_t OFF_BP1  = OFF_P16  + (size_t)NNODES * HSTRIDE * 2;       // N*160 f16
static const size_t OFF_BP2  = OFF_BP1  + (size_t)5 * 2 * 512 * 2;            // packed B frags
static const size_t OFF_BP3  = OFF_BP2  + (size_t)1 * 2 * 512 * 2;
static const size_t OFF_BP4  = OFF_BP3  + (size_t)1 * 2 * 512 * 2;

// ---------------- small utility kernels ----------------
__global__ void zero_f32_kernel(float* __restrict__ p, long long n) {
  long long i = (long long)blockIdx.x * blockDim.x + threadIdx.x;
  if (i < n) p[i] = 0.f;
}

// per-edge: ppr value + e2n scatter (both directions get edge_feat)
__global__ void edge_agg_kernel(const float* __restrict__ edge_feat,
                                const float* __restrict__ noise_mask,
                                const int*   __restrict__ edges,
                                float* __restrict__ ppr,
                                float* __restrict__ e2n) {
  int e = blockIdx.x * blockDim.x + threadIdx.x;
  if (e >= NEDGES) return;
  float mv = 0.f;
  #pragma unroll
  for (int k = 0; k < FEDGE; ++k) mv += edge_feat[(size_t)e * FEDGE + k];
  mv *= (1.f / (float)FEDGE);
  float nm = noise_mask[e];
  ppr[e] = 1.f - nm + nm * mv;
  int u = edges[2 * e + 0];
  int v = edges[2 * e + 1];
  #pragma unroll
  for (int k = 0; k < FEDGE; ++k) {
    float f = edge_feat[(size_t)e * FEDGE + k];
    atomicAdd(&e2n[(size_t)u * FEDGE + k], f);
    atomicAdd(&e2n[(size_t)v * FEDGE + k], f);
  }
}

// h0 = [node_feat(128) | e2n(6) | zeros(26)]  stride 160
__global__ void build_h0_kernel(const float* __restrict__ node_feat,
                                const float* __restrict__ e2n,
                                float* __restrict__ H) {
  long long tid = (long long)blockIdx.x * blockDim.x + threadIdx.x;
  if (tid >= (long long)NNODES * HSTRIDE) return;
  int n = (int)(tid / HSTRIDE);
  int d = (int)(tid - (long long)n * HSTRIDE);
  float val = 0.f;
  if (d < FNODE)              val = node_feat[(size_t)n * FNODE + d];
  else if (d < FNODE + FEDGE) val = e2n[(size_t)n * FEDGE + (d - FNODE)];
  H[tid] = val;
}

// Pack W (Kdim x outD f32) into per-lane WMMA B-fragment order:
//   Bp[((kc*nT + t)*32 + lane)*16 + j] = W[kc*32 + (lane>>4)*8 + (j&7) + (j>>3)*16][t*16 + (lane&15)]
__global__ void pack_b_kernel(const float* __restrict__ W, _Float16* __restrict__ Bp,
                              int Kdim, int outD, int nT, int total) {
  int tid = blockIdx.x * blockDim.x + threadIdx.x;
  if (tid >= total) return;
  int j     = tid & 15;
  int lane  = (tid >> 4) & 31;
  int chunk = tid >> 9;
  int t  = chunk % nT;
  int kc = chunk / nT;
  int hi = lane >> 4, colL = lane & 15;
  int col = t * 16 + colL;
  int K   = kc * 32 + hi * 8 + (j & 7) + (j >> 3) * 16;
  Bp[tid] = (K < Kdim && col < outD) ? (_Float16)W[(size_t)K * outD + col]
                                     : (_Float16)0.f;
}

// Pool = Hcur (first Kpad columns)
__global__ void copy_pool_kernel(const float* __restrict__ H, float* __restrict__ Pool,
                                 int Kpad) {
  long long tid = (long long)blockIdx.x * blockDim.x + threadIdx.x;
  if (tid >= (long long)NNODES * Kpad) return;
  int n = (int)(tid / Kpad);
  int d = (int)(tid - (long long)n * Kpad);
  Pool[(size_t)n * HSTRIDE + d] = H[(size_t)n * HSTRIDE + d];
}

// Pool16 = (f16)Pool  (first Kpad columns, stride HSTRIDE)
__global__ void cvt_pool_kernel(const float* __restrict__ Pool,
                                _Float16* __restrict__ P16, int Kpad) {
  long long tid = (long long)blockIdx.x * blockDim.x + threadIdx.x;
  if (tid >= (long long)NNODES * Kpad) return;
  int n = (int)(tid / Kpad);
  int d = (int)(tid - (long long)n * Kpad);
  P16[(size_t)n * HSTRIDE + d] = (_Float16)Pool[(size_t)n * HSTRIDE + d];
}

// Pool[u] += ppr*H[v];  Pool[v] += ppr*H[u]   (both directions)
__global__ void spmm_kernel(const int* __restrict__ edges,
                            const float* __restrict__ ppr,
                            const float* __restrict__ H,
                            float* __restrict__ Pool, int D) {
  long long tid = (long long)blockIdx.x * blockDim.x + threadIdx.x;
  if (tid >= (long long)NEDGES * D) return;
  int e = (int)(tid / D);
  int d = (int)(tid - (long long)e * D);
  int u = edges[2 * e + 0];
  int v = edges[2 * e + 1];
  float p = ppr[e];
  atomicAdd(&Pool[(size_t)u * HSTRIDE + d], p * H[(size_t)v * HSTRIDE + d]);
  atomicAdd(&Pool[(size_t)v * HSTRIDE + d], p * H[(size_t)u * HSTRIDE + d]);
}

// ---------------- WMMA layer kernel (compile-time specialized) ----------------
// 256 threads = 8 waves; wave w owns rows [blockIdx.x*128 + w*16, +16).
// lin = Pool(16xKPAD) @ W(KPAD x OUTD) + b ; h = tanh(lin / max(||lin||,eps))
template <int KPAD, int OUTD, int COLB>
__global__ __launch_bounds__(256) void layer_mm_kernel(
    const _Float16* __restrict__ P16,
    const _Float16* __restrict__ Bpack,
    const float*    __restrict__ bias,
    float* __restrict__ Hout,
    float* __restrict__ msg)
{
  constexpr int NK = KPAD / 32;
  constexpr int NT = (OUTD + 15) / 16;
  constexpr int BELEMS = NK * NT * 512;        // halves of packed B

  const int lane = threadIdx.x & 31;
  const int wave = threadIdx.x >> 5;
  const int rowBase = blockIdx.x * 128 + wave * 16;
  const int hi   = lane >> 4;                  // 0/1 lane half
  const int colL = lane & 15;

#if HAVE_ASYNC_LDS
  // Stage packed B fragments into LDS with async copy (shared by all 8 waves).
  __shared__ alignas(32) _Float16 Bs[BELEMS];
  for (int off = threadIdx.x * 8; off < BELEMS; off += 256 * 8) {
    __builtin_amdgcn_global_load_async_to_lds_b128(
        (gv4i_ptr)(Bpack + off), (lv4i_ptr)(Bs + off), 0, 0);
  }
  __builtin_amdgcn_s_wait_asynccnt(0);
  __syncthreads();
  const _Float16* __restrict__ Bsrc = (const _Float16*)Bs;
#else
  const _Float16* __restrict__ Bsrc = Bpack;
#endif

  // A layout: row = lane%16; K groups {base..base+7} and {base+16..base+23}, base=hi*8
  const _Float16* __restrict__ Ap = P16 + (size_t)(rowBase + colL) * HSTRIDE;
  __builtin_prefetch(Ap, 0, 1);                // global_prefetch_b8

  v8f acc0 = {};
  v8f acc1 = {};
  #pragma unroll
  for (int kc = 0; kc < NK; ++kc) {
    const int k0 = kc * 32 + hi * 8;
    v8h alo = *(const v8h*)(Ap + k0);
    v8h ahi = *(const v8h*)(Ap + k0 + 16);
    v16h a = __builtin_shufflevector(alo, ahi, 0, 1, 2, 3, 4, 5, 6, 7,
                                     8, 9, 10, 11, 12, 13, 14, 15);
    v16h b0 = *(const v16h*)(Bsrc + ((size_t)(kc * NT + 0) * 32 + lane) * 16);
    acc0 = __builtin_amdgcn_wmma_f32_16x16x32_f16(false, a, false, b0,
                                                  (short)0, acc0, false, false);
    if (NT > 1) {
      v16h b1 = *(const v16h*)(Bsrc + ((size_t)(kc * NT + 1) * 32 + lane) * 16);
      acc1 = __builtin_amdgcn_wmma_f32_16x16x32_f16(false, a, false, b1,
                                                    (short)0, acc1, false, false);
    }
  }

  // bias + row-norm + tanh.  C layout: M = r + 8*hi, N = colL (+16 for tile1).
  float bias0 = (colL < OUTD) ? bias[colL] : 0.f;
  float bias1 = (NT > 1 && 16 + colL < OUTD) ? bias[16 + colL] : 0.f;
  float c0[8], c1[8], ss[8];
  #pragma unroll
  for (int r = 0; r < 8; ++r) {
    c0[r] = (colL < OUTD) ? (acc0[r] + bias0) : 0.f;
    c1[r] = (NT > 1 && (16 + colL) < OUTD) ? (acc1[r] + bias1) : 0.f;
    ss[r] = c0[r] * c0[r] + c1[r] * c1[r];
  }
  #pragma unroll
  for (int r = 0; r < 8; ++r) {
    float s = ss[r];
    #pragma unroll
    for (int off = 1; off < 16; off <<= 1) s += __shfl_xor(s, off, 32);
    ss[r] = s;                        // row sum-of-squares (16 lanes per half)
  }
  const int mrow = rowBase + hi * 8;
  #pragma unroll
  for (int r = 0; r < 8; ++r) {
    float nrm  = fmaxf(sqrtf(ss[r]), 1e-12f);
    int   grow = mrow + r;
    if (colL < OUTD) {
      float h0 = tanhf(c0[r] / nrm);
      Hout[(size_t)grow * HSTRIDE + colL]      = h0;
      msg[(size_t)grow * DTOT + COLB + colL]   = h0;
    }
    if (NT > 1 && (16 + colL) < OUTD) {
      float h1 = tanhf(c1[r] / nrm);
      Hout[(size_t)grow * HSTRIDE + 16 + colL]     = h1;
      msg[(size_t)grow * DTOT + COLB + 16 + colL]  = h1;
    }
  }
}

// ---------------- top-k + conv head, one block per graph ----------------
__global__ __launch_bounds__(256) void head_kernel(
    const float* __restrict__ msg,
    const float* __restrict__ ck1, const float* __restrict__ cb1,
    const float* __restrict__ ck2, const float* __restrict__ cb2,
    const float* __restrict__ outW, const float* __restrict__ outb,
    float* __restrict__ out)
{
  const int g   = blockIdx.x;
  const int tid = threadIdx.x;

  __shared__ float s[NPGRAPH];
  __shared__ float rv[256];
  __shared__ int   ri[256];
  __shared__ int   idxs[KTOP];
  __shared__ float pooled[KTOP * DTOT];
  __shared__ float o1[C1N * KTOP];
  __shared__ float mp[C1N * (KTOP / 2)];
  __shared__ float o2[C2N * T2LEN];

  // scores = last channel of msg
  for (int i = tid; i < NPGRAPH; i += blockDim.x)
    s[i] = msg[((size_t)g * NPGRAPH + i) * DTOT + (DTOT - 1)];
  __syncthreads();

  // top-30 by iterated argmax (ties -> lower index, like lax.top_k)
  for (int k = 0; k < KTOP; ++k) {
    float bv = -3.402823466e+38f; int bi = 0x7fffffff;
    for (int i = tid; i < NPGRAPH; i += blockDim.x) {
      float v = s[i];
      if (v > bv || (v == bv && i < bi)) { bv = v; bi = i; }
    }
    rv[tid] = bv; ri[tid] = bi;
    __syncthreads();
    for (int off = 128; off > 0; off >>= 1) {
      if (tid < off) {
        if (rv[tid + off] > rv[tid] ||
            (rv[tid + off] == rv[tid] && ri[tid + off] < ri[tid])) {
          rv[tid] = rv[tid + off]; ri[tid] = ri[tid + off];
        }
      }
      __syncthreads();
    }
    if (tid == 0) { idxs[k] = ri[0]; s[ri[0]] = -3.402823466e+38f; }
    __syncthreads();
  }
  // sort indices ascending
  if (tid == 0) {
    for (int i = 1; i < KTOP; ++i) {
      int key = idxs[i]; int j = i - 1;
      while (j >= 0 && idxs[j] > key) { idxs[j + 1] = idxs[j]; --j; }
      idxs[j + 1] = key;
    }
  }
  __syncthreads();

  // gather pooled rows (30 x 97)
  for (int j = tid; j < KTOP * DTOT; j += blockDim.x) {
    int t = j / DTOT, d = j - t * DTOT;
    pooled[j] = msg[((size_t)g * NPGRAPH + idxs[t]) * DTOT + d];
  }
  __syncthreads();

  // conv1 (stride 97, kernel 97) + relu
  for (int j = tid; j < C1N * KTOP; j += blockDim.x) {
    int c = j / KTOP, t = j - c * KTOP;
    float acc = cb1[c];
    for (int d = 0; d < DTOT; ++d) acc += ck1[(size_t)c * DTOT + d] * pooled[t * DTOT + d];
    o1[c * KTOP + t] = fmaxf(acc, 0.f);
  }
  __syncthreads();

  // pairwise max-pool along time
  for (int j = tid; j < C1N * (KTOP / 2); j += blockDim.x) {
    int c = j / (KTOP / 2), t = j - c * (KTOP / 2);
    mp[c * (KTOP / 2) + t] = fmaxf(o1[c * KTOP + 2 * t], o1[c * KTOP + 2 * t + 1]);
  }
  __syncthreads();

  // conv2 (16->32 ch, width 5) + relu
  for (int j = tid; j < C2N * T2LEN; j += blockDim.x) {
    int c2 = j / T2LEN, t = j - c2 * T2LEN;
    float acc = cb2[c2];
    for (int c1 = 0; c1 < C1N; ++c1)
      #pragma unroll
      for (int kw = 0; kw < KW2C; ++kw)
        acc += ck2[((size_t)c2 * C1N + c1) * KW2C + kw] * mp[c1 * (KTOP / 2) + t + kw];
    o2[c2 * T2LEN + t] = fmaxf(acc, 0.f);
  }
  __syncthreads();

  // dense 352 -> 128 + relu
  for (int o = tid; o < OUTDIM; o += blockDim.x) {
    float acc = outb[o];
    for (int i = 0; i < DENSEIN; ++i) acc += o2[i] * outW[(size_t)i * OUTDIM + o];
    out[(size_t)g * OUTDIM + o] = fmaxf(acc, 0.f);
  }
}

// ---------------- launch ----------------
extern "C" void kernel_launch(void* const* d_in, const int* in_sizes, int n_in,
                              void* d_out, int out_size, void* d_ws, size_t ws_size,
                              hipStream_t stream) {
  const float* node_feat  = (const float*)d_in[0];
  const float* edge_feat  = (const float*)d_in[1];
  const float* noise_mask = (const float*)d_in[2];
  const int*   edges      = (const int*)  d_in[3];
  const float* W[4] = { (const float*)d_in[4], (const float*)d_in[6],
                        (const float*)d_in[8], (const float*)d_in[10] };
  const float* b[4] = { (const float*)d_in[5], (const float*)d_in[7],
                        (const float*)d_in[9], (const float*)d_in[11] };
  const float* convk1 = (const float*)d_in[12];
  const float* convb1 = (const float*)d_in[13];
  const float* convk2 = (const float*)d_in[14];
  const float* convb2 = (const float*)d_in[15];
  const float* outW   = (const float*)d_in[16];
  const float* outb   = (const float*)d_in[17];
  float* out = (float*)d_out;

  char* ws = (char*)d_ws;
  float*    ppr  = (float*)(ws + OFF_PPR);
  float*    e2n  = (float*)(ws + OFF_E2N);
  float*    Hcur = (float*)(ws + OFF_H);
  float*    Pool = (float*)(ws + OFF_POOL);
  float*    msg  = (float*)(ws + OFF_MSG);
  _Float16* P16  = (_Float16*)(ws + OFF_P16);
  _Float16* Bp[4] = { (_Float16*)(ws + OFF_BP1), (_Float16*)(ws + OFF_BP2),
                      (_Float16*)(ws + OFF_BP3), (_Float16*)(ws + OFF_BP4) };

  const int Kdim[4] = {134, 32, 32, 32};
  const int Kpad[4] = {160, 32, 32, 32};
  const int outD[4] = { 32, 32, 32,  1};
  const int nT[4]   = {  2,  2,  2,  1};

  // e2n = 0
  {
    long long n = (long long)NNODES * FEDGE;
    zero_f32_kernel<<<(unsigned)((n + 255) / 256), 256, 0, stream>>>(e2n, n);
  }
  // edge prep
  edge_agg_kernel<<<(NEDGES + 255) / 256, 256, 0, stream>>>(
      edge_feat, noise_mask, edges, ppr, e2n);
  // h0
  {
    long long n = (long long)NNODES * HSTRIDE;
    build_h0_kernel<<<(unsigned)((n + 255) / 256), 256, 0, stream>>>(node_feat, e2n, Hcur);
  }
  // pack weights into f16 B-fragment order
  for (int l = 0; l < 4; ++l) {
    int total = (Kpad[l] / 32) * nT[l] * 512;
    pack_b_kernel<<<(total + 255) / 256, 256, 0, stream>>>(W[l], Bp[l],
                                                           Kdim[l], outD[l], nT[l], total);
  }

  // 4 GNN layers: Pool=H ; Pool+=SPMM ; Pool16=f16(Pool) ; WMMA matmul+norm+tanh
  for (int l = 0; l < 4; ++l) {
    long long nc = (long long)NNODES * Kpad[l];
    copy_pool_kernel<<<(unsigned)((nc + 255) / 256), 256, 0, stream>>>(Hcur, Pool, Kpad[l]);
    long long ns = (long long)NEDGES * Kdim[l];
    spmm_kernel<<<(unsigned)((ns + 255) / 256), 256, 0, stream>>>(edges, ppr, Hcur, Pool, Kdim[l]);
    cvt_pool_kernel<<<(unsigned)((nc + 255) / 256), 256, 0, stream>>>(Pool, P16, Kpad[l]);
    switch (l) {
      case 0: layer_mm_kernel<160, 32,  0><<<NNODES / 128, 256, 0, stream>>>(
                  P16, Bp[0], b[0], Hcur, msg); break;
      case 1: layer_mm_kernel< 32, 32, 32><<<NNODES / 128, 256, 0, stream>>>(
                  P16, Bp[1], b[1], Hcur, msg); break;
      case 2: layer_mm_kernel< 32, 32, 64><<<NNODES / 128, 256, 0, stream>>>(
                  P16, Bp[2], b[2], Hcur, msg); break;
      case 3: layer_mm_kernel< 32,  1, 96><<<NNODES / 128, 256, 0, stream>>>(
                  P16, Bp[3], b[3], Hcur, msg); break;
    }
  }
  // top-k + convs + dense head
  head_kernel<<<BGRAPH, 256, 0, stream>>>(msg, convk1, convb1, convk2, convb2,
                                          outW, outb, out);
}
